// BDH_GPU_20031727469015
// MI455X (gfx1250) — matrix operand order repaired and checked by reading, not compiled, not run on hardware
//
#include <hip/hip_runtime.h>

// ---------------------------------------------------------------------------
// BDH forward on MI455X (gfx1250, wave32, WMMA).
// - GEMMs: v_wmma_f32_16x16x32_bf16, fp32 accumulate.
// - Weight tiles async-staged to LDS (global_load_async_to_lds_b128),
//   double-buffered, retired with graduated s_wait_asynccnt.
// - sched_barrier(0) keeps fragment-load clauses ahead of WMMA chains.
// ---------------------------------------------------------------------------

#define DD    256
#define NN    8192
#define TT    2048
#define BB    8
#define BT    16384
#define MH    2048
#define NL    6
#define VOC   256
#define EPSL  1e-5f

typedef __bf16 v16bf __attribute__((ext_vector_type(16)));
typedef float  v8f   __attribute__((ext_vector_type(8)));

#if defined(__has_builtin)
#if __has_builtin(__builtin_amdgcn_sched_barrier)
#define SCHED_BARRIER() __builtin_amdgcn_sched_barrier(0)
#endif
#endif
#ifndef SCHED_BARRIER
#define SCHED_BARRIER()
#endif

#define V8F_ZERO v8f{0.f, 0.f, 0.f, 0.f, 0.f, 0.f, 0.f, 0.f}
#define WAIT_ASYNC(n) asm volatile("s_wait_asynccnt " #n ::: "memory")

union BF16x16 { v16bf v; uint4 q[2]; };

__device__ inline v8f wmma_bf16(v16bf a, v16bf b, v8f c) {
  return __builtin_amdgcn_wmma_f32_16x16x32_bf16(false, a, false, b,
                                                 (short)0, c, false, false);
}

// A fragment (16x32, bf16) from row-major matrix, leading dim ld (elements).
__device__ inline v16bf load_a_frag(const __bf16* base, int ld) {
  const int lane = threadIdx.x & 31;
  const int row  = lane & 15;
  const int kb   = (lane >> 4) << 3;
  const __bf16* p = base + row * ld + kb;
  BF16x16 u;
  u.q[0] = *reinterpret_cast<const uint4*>(p);
  u.q[1] = *reinterpret_cast<const uint4*>(p + 16);
  return u.v;
}

// B fragment (32x16, bf16) from k-contiguous layout: column j at baseT + j*ldk.
__device__ inline v16bf load_b_frag(const __bf16* baseT, int ldk) {
  const int lane = threadIdx.x & 31;
  const int col  = lane & 15;
  const int kb   = (lane >> 4) << 3;
  const __bf16* p = baseT + col * ldk + kb;
  BF16x16 u;
  u.q[0] = *reinterpret_cast<const uint4*>(p);
  u.q[1] = *reinterpret_cast<const uint4*>(p + 16);
  return u.v;
}

__device__ inline void store_c_tile(float* base, int ld, v8f acc) {
  const int lane = threadIdx.x & 31;
  const int col  = lane & 15;
  const int r0   = (lane >> 4) << 3;
#pragma unroll
  for (int r = 0; r < 8; ++r) base[(r0 + r) * ld + col] = acc[r];
}

__device__ inline float wred32(float x) {
  x += __shfl_xor(x, 16, 32);
  x += __shfl_xor(x, 8, 32);
  x += __shfl_xor(x, 4, 32);
  x += __shfl_xor(x, 2, 32);
  x += __shfl_xor(x, 1, 32);
  return x;
}

// Async global->LDS 16B copy (CDNA5 ASYNCcnt path); lane-granular.
__device__ inline void async_copy_b128(const __bf16* g, __bf16* l) {
  unsigned loff = (unsigned)(size_t)l;
  unsigned long long ga = (unsigned long long)(size_t)g;
  asm volatile("global_load_async_to_lds_b128 %0, %1, off"
               :: "v"(loff), "v"(ga) : "memory");
}
__device__ inline void wait_ds0() {
  asm volatile("s_wait_dscnt 0" ::: "memory");
}

// ------------------------- weight / table prep -----------------------------

__global__ void conv_decoder_kernel(const float* __restrict__ dx,
                                    __bf16* __restrict__ WT) {
  int i = blockIdx.x * 256 + threadIdx.x;       // over N*D
  int n = i >> 8, d = i & 255;
  int h = n >> 11, m = n & 2047;
  WT[i] = (__bf16)dx[(size_t)h * (DD * MH) + (size_t)d * MH + m];
}

__global__ void conv_enc_kernel(const float* __restrict__ enc,
                                __bf16* __restrict__ encT) {
  int i = blockIdx.x * 256 + threadIdx.x;       // over D*N
  int d = i >> 13, n = i & 8191;
  encT[i] = (__bf16)enc[(size_t)n * DD + d];
}

__global__ void conv_rd_kernel(const float* __restrict__ rd,
                               __bf16* __restrict__ rdT) {
  int i = blockIdx.x * 256 + threadIdx.x;       // over VOC*D
  int vo = i >> 8, d = i & 255;
  rdT[i] = (__bf16)rd[(size_t)d * VOC + vo];
}

__global__ void tables_kernel(float* __restrict__ cosT, float* __restrict__ sinT) {
  int i = blockIdx.x * 256 + threadIdx.x;       // over T*(D/2)
  int t = i >> 7, j = i & 127;
  float invf = __expf(-((float)(2 * j) / (float)DD) * 9.2103403719761836f);
  float s, c;
  __sincosf((float)t * invf, &s, &c);
  cosT[i] = c; sinT[i] = s;
}

// ------------------------- row-wise (LN-family) ----------------------------

__global__ void __launch_bounds__(256) embed_ln_kernel(
    const int* __restrict__ idx, const float* __restrict__ wte,
    float* __restrict__ vf, __bf16* __restrict__ vb) {
  const int wv = threadIdx.x >> 5, lane = threadIdx.x & 31;
  const int row = blockIdx.x * 8 + wv;
  const float* e = wte + (size_t)idx[row] * DD;
  float x[8], s = 0.f;
#pragma unroll
  for (int i = 0; i < 8; ++i) { x[i] = e[i * 32 + lane]; s += x[i]; }
  float mu = wred32(s) * (1.f / DD);
  float q = 0.f;
#pragma unroll
  for (int i = 0; i < 8; ++i) { float d = x[i] - mu; q += d * d; }
  float inv = rsqrtf(wred32(q) * (1.f / DD) + EPSL);
  float* o = vf + (size_t)row * DD;
  __bf16* ob = vb + (size_t)row * DD;
#pragma unroll
  for (int i = 0; i < 8; ++i) {
    float y = (x[i] - mu) * inv;
    o[i * 32 + lane] = y;
    ob[i * 32 + lane] = (__bf16)y;
  }
}

__global__ void __launch_bounds__(256) ln_a_kernel(
    const float* __restrict__ a, __bf16* __restrict__ lna) {
  const int wv = threadIdx.x >> 5, lane = threadIdx.x & 31;
  const int row = blockIdx.x * 8 + wv;
  const float* e = a + (size_t)row * DD;
  float x[8], s = 0.f;
#pragma unroll
  for (int i = 0; i < 8; ++i) { x[i] = e[i * 32 + lane]; s += x[i]; }
  float mu = wred32(s) * (1.f / DD);
  float q = 0.f;
#pragma unroll
  for (int i = 0; i < 8; ++i) { float d = x[i] - mu; q += d * d; }
  float inv = rsqrtf(wred32(q) * (1.f / DD) + EPSL);
  __bf16* ob = lna + (size_t)row * DD;
#pragma unroll
  for (int i = 0; i < 8; ++i) ob[i * 32 + lane] = (__bf16)((x[i] - mu) * inv);
}

__global__ void __launch_bounds__(256) residual_kernel(
    const float* __restrict__ upd, float* __restrict__ vf,
    __bf16* __restrict__ vb) {
  const int wv = threadIdx.x >> 5, lane = threadIdx.x & 31;
  const int row = blockIdx.x * 8 + wv;
  const float* u = upd + (size_t)row * DD;
  float x[8], s = 0.f;
#pragma unroll
  for (int i = 0; i < 8; ++i) { x[i] = u[i * 32 + lane]; s += x[i]; }
  float mu = wred32(s) * (1.f / DD);
  float q = 0.f;
#pragma unroll
  for (int i = 0; i < 8; ++i) { float d = x[i] - mu; q += d * d; }
  float inv = rsqrtf(wred32(q) * (1.f / DD) + EPSL);
  float* vrow = vf + (size_t)row * DD;
  float y[8]; s = 0.f;
#pragma unroll
  for (int i = 0; i < 8; ++i) {
    y[i] = vrow[i * 32 + lane] + (x[i] - mu) * inv;
    s += y[i];
  }
  mu = wred32(s) * (1.f / DD);
  q = 0.f;
#pragma unroll
  for (int i = 0; i < 8; ++i) { float d = y[i] - mu; q += d * d; }
  inv = rsqrtf(wred32(q) * (1.f / DD) + EPSL);
  __bf16* ob = vb + (size_t)row * DD;
#pragma unroll
  for (int i = 0; i < 8; ++i) {
    float z = (y[i] - mu) * inv;
    vrow[i * 32 + lane] = z;
    ob[i * 32 + lane] = (__bf16)z;
  }
}

__global__ void __launch_bounds__(256) rope_kernel(
    const float* __restrict__ vf, const float* __restrict__ cosT,
    const float* __restrict__ sinT, __bf16* __restrict__ qr) {
  const int wv = threadIdx.x >> 5, lane = threadIdx.x & 31;
  const int row = blockIdx.x * 8 + wv;
  const int t = row & (TT - 1);
  const float* x = vf + (size_t)row * DD;
  __bf16* o = qr + (size_t)row * DD;
#pragma unroll
  for (int i = 0; i < 4; ++i) {
    int j = i * 32 + lane;                       // 0..127
    float c = cosT[t * 128 + j], sn = sinT[t * 128 + j];
    float x0 = x[j], x1 = x[j + 128];
    o[j]       = (__bf16)(x0 * c - x1 * sn);
    o[j + 128] = (__bf16)(x1 * c + x0 * sn);
  }
}

// ------------------------- attention (flash-linear) ------------------------
// LDS: vT [256][128] (transposed v chunk) | qrS [128][256] (async-staged raw
// key chunk) | Ebuf 8x[16][128] per-wave E scratch. 160KB dynamic.
__global__ void __launch_bounds__(256) attn_kernel(
    const __bf16* __restrict__ qr, const __bf16* __restrict__ vbf,
    float* __restrict__ aout) {
  extern __shared__ char smem[];
  __bf16* vT   = (__bf16*)smem;
  __bf16* qrS  = (__bf16*)(smem + 65536);
  __bf16* Ebuf = (__bf16*)(smem + 131072);

  const int qc = blockIdx.x, b = blockIdx.y;
  const int wv = threadIdx.x >> 5, lane = threadIdx.x & 31;
  const int q0 = qc * 128;
  const int rowg = b * TT + q0 + wv * 16;

  v16bf aQ[8];
#pragma unroll
  for (int kk = 0; kk < 8; ++kk)
    aQ[kk] = load_a_frag(qr + (size_t)rowg * DD + kk * 32, DD);

  v8f accD[16];
#pragma unroll
  for (int i = 0; i < 16; ++i) accD[i] = V8F_ZERO;

  __bf16* Ew = Ebuf + wv * (16 * 128);
  const int tRow = q0 + wv * 16 + ((lane >> 4) << 3);

  for (int kc = 0; kc <= qc; ++kc) {
    __syncthreads();
    {
      // async raw copy of qr key chunk (4096 x 16B)
      const __bf16* qsrc = qr + (size_t)(b * TT + kc * 128) * DD;
      for (int i = threadIdx.x; i < (128 * DD) / 8; i += 256)
        async_copy_b128(qsrc + (size_t)i * 8, qrS + (size_t)i * 8);
      // manual transposed staging of v chunk: vT[d][s]
      const __bf16* vsrc = vbf + (size_t)(b * TT + kc * 128) * DD;
      for (int i = threadIdx.x; i < (128 * DD) / 8; i += 256) {
        int s = (i * 8) >> 8, d0 = (i * 8) & 255;
        uint4 pk = *reinterpret_cast<const uint4*>(vsrc + (size_t)i * 8);
        const __bf16* ph = reinterpret_cast<const __bf16*>(&pk);
#pragma unroll
        for (int e = 0; e < 8; ++e) vT[(d0 + e) * 128 + s] = ph[e];
      }
      WAIT_ASYNC(0);
    }
    __syncthreads();

    // energy tiles E[16 x 128] -> Ew (bf16)
#pragma unroll 1
    for (int jt = 0; jt < 8; ++jt) {
      const __bf16* kb = qrS + (size_t)(jt * 16) * DD;
      v16bf bq[8];
#pragma unroll
      for (int kk = 0; kk < 8; ++kk) bq[kk] = load_b_frag(kb + kk * 32, DD);
      SCHED_BARRIER();
      v8f e0 = V8F_ZERO, e1 = V8F_ZERO;
#pragma unroll
      for (int kk = 0; kk < 4; ++kk) {
        e0 = wmma_bf16(aQ[kk], bq[kk], e0);
        e1 = wmma_bf16(aQ[kk + 4], bq[kk + 4], e1);
      }
      v8f e = e0 + e1;
      if (kc == qc) {
        int s = kc * 128 + jt * 16 + (lane & 15);
#pragma unroll
        for (int r = 0; r < 8; ++r)
          if (s > tRow + r) e[r] = 0.f;
      }
      {
        int col = lane & 15, r0 = (lane >> 4) << 3;
#pragma unroll
        for (int r = 0; r < 8; ++r)
          Ew[(r0 + r) * 128 + jt * 16 + col] = (__bf16)e[r];
      }
    }
    wait_ds0();

    // accD += E(16x128) @ v_k(128x256)
#pragma unroll 1
    for (int ks = 0; ks < 4; ++ks) {
      v16bf af = load_a_frag(Ew + ks * 32, 128);
#pragma unroll
      for (int dtb = 0; dtb < 4; ++dtb) {
        v16bf bv[4];
#pragma unroll
        for (int j = 0; j < 4; ++j)
          bv[j] = load_b_frag(vT + (size_t)(dtb * 4 + j) * 16 * 128 + ks * 32, 128);
        SCHED_BARRIER();
#pragma unroll
        for (int j = 0; j < 4; ++j)
          accD[dtb * 4 + j] = wmma_bf16(af, bv[j], accD[dtb * 4 + j]);
      }
    }
  }

  float* ob = aout + (size_t)rowg * DD;
#pragma unroll
  for (int dt = 0; dt < 16; ++dt) store_c_tile(ob + dt * 16, DD, accD[dt]);
}

// ------------------------- fused decoder/encoder ---------------------------
// upd = (relu(lna@Wy) * relu(v@Wx)) @ enc over 64-wide N tiles.
// Per-tile Wx/Wy/enc slabs (3 x 32KB) async-staged to LDS, double buffered:
// issue tile nt+1, s_wait_asynccnt 24 retires tile nt, barrier, compute.
// LDS pointers are computed arithmetically (no LDS pointer arrays -> no
// addrspacecast static initializers).
__global__ void __launch_bounds__(256) fused_decoder_kernel(
    const __bf16* __restrict__ vbf, const __bf16* __restrict__ lna,
    const __bf16* __restrict__ WxT, const __bf16* __restrict__ WyT,
    const __bf16* __restrict__ encT, float* __restrict__ upd) {
  extern __shared__ char smem[];
  // layout: [wx0|wx1|wy0|wy1|en0|en1|Ybuf], each weight slab 32KB
  __bf16* Ybuf = (__bf16*)(smem + 196608);     // 8 x [16][64]

  const int wv = threadIdx.x >> 5, lane = threadIdx.x & 31;
  const int rowg = blockIdx.x * 128 + wv * 16;

  auto stage = [&](int nt, int bi) {
    const int n0 = nt * 64;
    const __bf16* sx = WxT + (size_t)n0 * DD;    // contiguous 64x256 slab
    const __bf16* sy = WyT + (size_t)n0 * DD;
    __bf16* wx = (__bf16*)(smem + (size_t)bi * 32768);
    __bf16* wy = (__bf16*)(smem + 65536 + (size_t)bi * 32768);
    __bf16* en = (__bf16*)(smem + 131072 + (size_t)bi * 32768);
    for (int i = threadIdx.x; i < 2048; i += 256) {   // 8 iters x 3 = 24 ops
      async_copy_b128(sx + (size_t)i * 8, wx + (size_t)i * 8);
      async_copy_b128(sy + (size_t)i * 8, wy + (size_t)i * 8);
      int d = i >> 3, seg = i & 7;                    // encT[d][n0+seg*8..]
      async_copy_b128(encT + (size_t)d * NN + n0 + seg * 8, en + d * 64 + seg * 8);
    }
  };

  v16bf aV[8], aL[8];
#pragma unroll
  for (int kk = 0; kk < 8; ++kk) {
    aV[kk] = load_a_frag(vbf + (size_t)rowg * DD + kk * 32, DD);
    aL[kk] = load_a_frag(lna + (size_t)rowg * DD + kk * 32, DD);
  }
  v8f accU[16];
#pragma unroll
  for (int i = 0; i < 16; ++i) accU[i] = V8F_ZERO;

  __bf16* Yw = Ybuf + wv * (16 * 64);

  stage(0, 0);
  for (int nt = 0; nt < NN / 64; ++nt) {
    const int cur = nt & 1;
    if (nt + 1 < NN / 64) {
      stage(nt + 1, cur ^ 1);
      WAIT_ASYNC(24);           // retire tile nt (in-order completion)
    } else {
      WAIT_ASYNC(0);
    }
    __syncthreads();
    __bf16* wx = (__bf16*)(smem + (size_t)cur * 32768);
    __bf16* wy = (__bf16*)(smem + 65536 + (size_t)cur * 32768);
    __bf16* en = (__bf16*)(smem + 131072 + (size_t)cur * 32768);

#pragma unroll 1
    for (int jt = 0; jt < 4; ++jt) {
      v16bf bw[8];
#pragma unroll
      for (int kk = 0; kk < 8; ++kk)
        bw[kk] = load_b_frag(wx + (size_t)(jt * 16) * DD + kk * 32, DD);
      SCHED_BARRIER();
      v8f ax0 = V8F_ZERO, ax1 = V8F_ZERO;
#pragma unroll
      for (int kk = 0; kk < 4; ++kk) {
        ax0 = wmma_bf16(aV[kk], bw[kk], ax0);
        ax1 = wmma_bf16(aV[kk + 4], bw[kk + 4], ax1);
      }
#pragma unroll
      for (int kk = 0; kk < 8; ++kk)
        bw[kk] = load_b_frag(wy + (size_t)(jt * 16) * DD + kk * 32, DD);
      SCHED_BARRIER();
      v8f ay0 = V8F_ZERO, ay1 = V8F_ZERO;
#pragma unroll
      for (int kk = 0; kk < 4; ++kk) {
        ay0 = wmma_bf16(aL[kk], bw[kk], ay0);
        ay1 = wmma_bf16(aL[kk + 4], bw[kk + 4], ay1);
      }
      v8f ax = ax0 + ax1, ay = ay0 + ay1;
      int col = lane & 15, r0 = (lane >> 4) << 3;
#pragma unroll
      for (int r = 0; r < 8; ++r) {
        float yv = fmaxf(ay[r], 0.f) * fmaxf(ax[r], 0.f);
        Yw[(r0 + r) * 64 + jt * 16 + col] = (__bf16)yv;
      }
    }
    wait_ds0();
#pragma unroll 1
    for (int ks = 0; ks < 2; ++ks) {
      v16bf af = load_a_frag(Yw + ks * 32, 64);
#pragma unroll
      for (int half = 0; half < 2; ++half) {
        v16bf be[8];
#pragma unroll
        for (int j = 0; j < 8; ++j)
          be[j] = load_b_frag(en + (size_t)((half * 8 + j) * 16) * 64 + ks * 32, 64);
        SCHED_BARRIER();
#pragma unroll
        for (int j = 0; j < 8; ++j)
          accU[half * 8 + j] = wmma_bf16(af, be[j], accU[half * 8 + j]);
      }
    }
    __syncthreads();
  }
  float* ob = upd + (size_t)rowg * DD;
#pragma unroll
  for (int dt = 0; dt < 16; ++dt) store_c_tile(ob + dt * 16, DD, accU[dt]);
}

// ------------------------- readout -----------------------------------------
__global__ void __launch_bounds__(256) logits_kernel(
    const __bf16* __restrict__ vbf, const __bf16* __restrict__ rdT,
    float* __restrict__ out) {
  const int wv = threadIdx.x >> 5;
  const int rowg = blockIdx.x * 128 + wv * 16;
  v16bf aV[8];
#pragma unroll
  for (int kk = 0; kk < 8; ++kk)
    aV[kk] = load_a_frag(vbf + (size_t)rowg * DD + kk * 32, DD);
#pragma unroll 1
  for (int vt = 0; vt < 16; ++vt) {
    v16bf br[8];
#pragma unroll
    for (int kk = 0; kk < 8; ++kk)
      br[kk] = load_b_frag(rdT + (size_t)(vt * 16) * DD + kk * 32, DD);
    SCHED_BARRIER();
    v8f a0 = V8F_ZERO, a1 = V8F_ZERO;
#pragma unroll
    for (int kk = 0; kk < 4; ++kk) {
      a0 = wmma_bf16(aV[kk], br[kk], a0);
      a1 = wmma_bf16(aV[kk + 4], br[kk + 4], a1);
    }
    store_c_tile(out + (size_t)rowg * VOC + vt * 16, VOC, a0 + a1);
  }
}

// ------------------------- launch ------------------------------------------
extern "C" void kernel_launch(void* const* d_in, const int* in_sizes, int n_in,
                              void* d_out, int out_size, void* d_ws,
                              size_t ws_size, hipStream_t stream) {
  (void)in_sizes; (void)n_in; (void)out_size; (void)ws_size;
  const int*   idx = (const int*)d_in[0];
  const float* wte = (const float*)d_in[1];
  const float* enc = (const float*)d_in[2];
  const float* dx  = (const float*)d_in[3];
  const float* dy  = (const float*)d_in[4];
  const float* rd  = (const float*)d_in[5];
  float* out = (float*)d_out;

  char* ws = (char*)d_ws;
  size_t off = 0;
  auto alloc = [&](size_t bytes) -> void* {
    void* p = ws + off;
    off += (bytes + 255) & ~(size_t)255;
    return p;
  };
  float*  vf   = (float*)alloc((size_t)BT * DD * 4);
  __bf16* vb   = (__bf16*)alloc((size_t)BT * DD * 2);
  __bf16* qr   = (__bf16*)alloc((size_t)BT * DD * 2);
  float*  aupd = (float*)alloc((size_t)BT * DD * 4);  // a, then upd
  __bf16* lna  = (__bf16*)alloc((size_t)BT * DD * 2);
  __bf16* WxT  = (__bf16*)alloc((size_t)NN * DD * 2);
  __bf16* WyT  = (__bf16*)alloc((size_t)NN * DD * 2);
  __bf16* encT = (__bf16*)alloc((size_t)DD * NN * 2);
  __bf16* rdT  = (__bf16*)alloc((size_t)VOC * DD * 2);
  float*  cosT = (float*)alloc((size_t)TT * (DD / 2) * 4);
  float*  sinT = (float*)alloc((size_t)TT * (DD / 2) * 4);

  conv_decoder_kernel<<<NN * DD / 256, 256, 0, stream>>>(dx, WxT);
  conv_decoder_kernel<<<NN * DD / 256, 256, 0, stream>>>(dy, WyT);
  conv_enc_kernel<<<DD * NN / 256, 256, 0, stream>>>(enc, encT);
  conv_rd_kernel<<<VOC * DD / 256, 256, 0, stream>>>(rd, rdT);
  tables_kernel<<<TT * (DD / 2) / 256, 256, 0, stream>>>(cosT, sinT);

  embed_ln_kernel<<<BT / 8, 256, 0, stream>>>(idx, wte, vf, vb);

  const size_t attn_smem  = 65536 + 65536 + 32768;          // 160 KB
  const size_t fused_smem = 6 * 32768 + 8 * 16 * 64 * 2;    // 208 KB
  for (int l = 0; l < NL; ++l) {
    rope_kernel<<<BT / 8, 256, 0, stream>>>(vf, cosT, sinT, qr);
    attn_kernel<<<dim3(TT / 128, BB), 256, attn_smem, stream>>>(qr, vb, aupd);
    ln_a_kernel<<<BT / 8, 256, 0, stream>>>(aupd, lna);
    fused_decoder_kernel<<<BT / 128, 256, fused_smem, stream>>>(
        vb, lna, WxT, WyT, encT, aupd);
    residual_kernel<<<BT / 8, 256, 0, stream>>>(aupd, vf, vb);
  }
  logits_kernel<<<BT / 128, 256, 0, stream>>>(vb, rdT, out);
}
